// AttentionBlock_19834158973225
// MI455X (gfx1250) — matrix operand order repaired and checked by reading, not compiled, not run on hardware
//
#include <hip/hip_runtime.h>
#include <hip/hip_bf16.h>
#include <stdint.h>

// ---------------------------------------------------------------------------
// AttentionBlock on MI455X (gfx1250).
// Algebraic reduction: softmax over its own reduced axis sums to 1, so the
// attention is the identity on V.  out = proj@(Wv@GN(x)+bv) + proj_b + x
// collapses to one per-batch 512x512xHW GEMM on raw x with a folded matrix.
// GEMM runs as bf16x3 split (hi*hi + hi*lo + lo*hi) on v_wmma_f32_16x16x32_bf16
// with f32 accumulators -> near-fp32 accuracy at bf16 matrix-core rates.
// A' weight tiles are staged with global_load_async_to_lds_b128 (ASYNCcnt,
// double-buffered); x tiles transit VGPRs (batched loads) for the bf16 split.
// ---------------------------------------------------------------------------

typedef __attribute__((ext_vector_type(8)))  float  v8f;
typedef __attribute__((ext_vector_type(16))) __bf16 v16bf;

#define WMMA_BF16(A, Bm, Cm) \
  __builtin_amdgcn_wmma_f32_16x16x32_bf16(false, (A), false, (Bm), (short)0, (Cm), false, false)

static constexpr int CC  = 512;    // channels
static constexpr int HW  = 4096;   // spatial
static constexpr int NB  = 16;     // batch
static constexpr int NG  = 8;      // groups
static constexpr int LDSS = 40;    // LDS row stride in halfs (80B, 16B aligned, padded)

__device__ __forceinline__ unsigned short f32_to_bf16_rne(float f) {
  unsigned int u = __float_as_uint(f);
  u += 0x7FFFu + ((u >> 16) & 1u);          // round-to-nearest-even
  return (unsigned short)(u >> 16);
}

__device__ __forceinline__ unsigned lds_off_u32(const void* p) {
  // flat LDS pointers carry the byte offset in the low 32 bits (aperture in high)
  return (unsigned)(uintptr_t)p;
}

// --------------------------- K1: GroupNorm stats ---------------------------
__global__ __launch_bounds__(256) void gn_stats_kernel(
    const float* __restrict__ x, float* __restrict__ mean, float* __restrict__ rstd) {
  __shared__ float ss[256], sq[256];
  const int bg = blockIdx.x;                          // b*8+g, matches memory layout
  const size_t base = (size_t)bg * (CC / NG) * HW;    // contiguous 64*4096 block
  float s = 0.f, q = 0.f;
  for (int i = threadIdx.x; i < (CC / NG) * HW; i += 256) {
    float v = x[base + i];
    s += v; q += v * v;
  }
  ss[threadIdx.x] = s; sq[threadIdx.x] = q;
  __syncthreads();
  for (int st = 128; st > 0; st >>= 1) {
    if (threadIdx.x < st) { ss[threadIdx.x] += ss[threadIdx.x + st]; sq[threadIdx.x] += sq[threadIdx.x + st]; }
    __syncthreads();
  }
  if (threadIdx.x == 0) {
    const float invN = 1.f / (float)((CC / NG) * HW);
    float m = ss[0] * invN;
    float var = sq[0] * invN - m * m;
    if (var < 0.f) var = 0.f;
    mean[bg] = m;
    rstd[bg] = rsqrtf(var + 1e-5f);
  }
}

// ------------------ K2: M = proj_w @ Wv  (+ bv bias fold) ------------------
__global__ __launch_bounds__(256) void make_M_kernel(
    const float* __restrict__ qkv_w, const float* __restrict__ proj_w,
    const float* __restrict__ qkv_b, float* __restrict__ M, float* __restrict__ pb2) {
  const int o0 = blockIdx.x * 8;
  const int c0 = threadIdx.x;
  float a0[8] = {}, a1[8] = {};
  for (int t = 0; t < CC; ++t) {
    const int vr = (t >> 7) * 384 + 256 + (t & 127);   // v-rows of qkv_w
    const float* row = qkv_w + (size_t)vr * CC;
    const float r0 = row[c0], r1 = row[c0 + 256];
#pragma unroll
    for (int j = 0; j < 8; ++j) {
      const float w = proj_w[(size_t)(o0 + j) * CC + t];
      a0[j] += w * r0; a1[j] += w * r1;
    }
  }
#pragma unroll
  for (int j = 0; j < 8; ++j) {
    M[(size_t)(o0 + j) * CC + c0]       = a0[j];
    M[(size_t)(o0 + j) * CC + c0 + 256] = a1[j];
  }
  if (threadIdx.x < 8) {               // proj_w @ bv  (batch-independent bias term)
    const int o = o0 + threadIdx.x;
    float s = 0.f;
    for (int t = 0; t < CC; ++t) {
      const int vr = (t >> 7) * 384 + 256 + (t & 127);
      s += proj_w[(size_t)o * CC + t] * qkv_b[vr];
    }
    pb2[o] = s;
  }
}

// ------- K3: per-batch fold A'[b]=M*a[b,c] (split bf16 hi/lo) + bias' -------
__global__ __launch_bounds__(256) void fold_M_kernel(
    const float* __restrict__ M, const float* __restrict__ mean, const float* __restrict__ rstd,
    const float* __restrict__ gamma, const float* __restrict__ beta,
    const float* __restrict__ proj_b, const float* __restrict__ pb2,
    unsigned short* __restrict__ Ahi, unsigned short* __restrict__ Alo,
    float* __restrict__ biasP) {
  __shared__ float red[256];
  const int o = blockIdx.x, b = blockIdx.y;
  const float* mrow = M + (size_t)o * CC;
  const size_t outBase = ((size_t)b * CC + o) * CC;
  float part = 0.f;
#pragma unroll
  for (int i = 0; i < 2; ++i) {
    const int c = threadIdx.x + i * 256;
    const int grp = c >> 6;
    const float rs = rstd[b * NG + grp];
    const float a  = rs * gamma[c];
    const float d  = beta[c] - mean[b * NG + grp] * a;
    const float mv = mrow[c];
    const float ap = mv * a;
    const unsigned short hb = f32_to_bf16_rne(ap);
    const float lof = ap - __uint_as_float((unsigned int)hb << 16);
    Ahi[outBase + c] = hb;
    Alo[outBase + c] = f32_to_bf16_rne(lof);
    part += mv * d;                    // bias term kept in full fp32
  }
  red[threadIdx.x] = part;
  __syncthreads();
  for (int st = 128; st > 0; st >>= 1) {
    if (threadIdx.x < st) red[threadIdx.x] += red[threadIdx.x + st];
    __syncthreads();
  }
  if (threadIdx.x == 0)
    biasP[(size_t)b * CC + o] = red[0] + proj_b[o] + pb2[o];
}

// ------------------- K4: main GEMM  out = A'@x + bias' + x ------------------
// Block: 128x128 output tile, 8 waves (wave32) as 4x2 -> 32x64 per wave,
// BK=32 (one WMMA K-slab), 24 v_wmma per K step per wave (bf16x3).
// A' tiles: async global->LDS copies, double-buffered on ASYNCcnt.
union V16 { uint4 q[2]; v16bf v; };

__global__ __launch_bounds__(256) void attn_gemm_kernel(
    const float* __restrict__ x, const unsigned short* __restrict__ Ahi,
    const unsigned short* __restrict__ Alo, const float* __restrict__ biasP,
    float* __restrict__ out) {
  __shared__ unsigned short sA[2][2][128 * LDSS];   // [buffer][hi/lo][row*LDSS+k]
  __shared__ unsigned short sBhi[128 * LDSS];       // stored transposed: [col][k]
  __shared__ unsigned short sBlo[128 * LDSS];
  __shared__ float sBias[128];

  const int tid  = threadIdx.x;
  const int lane = tid & 31;
  const int wave = tid >> 5;
  const int wr = wave & 3;        // wave row block (32 rows)
  const int wc = wave >> 2;       // wave col block (64 cols)
  const int lg = lane >> 4;       // lane group (0/1)
  const int lm = lane & 15;

  const int b       = blockIdx.z;
  const int rowBase = blockIdx.y * 128;
  const int colBase = blockIdx.x * 128;
  const size_t xBatch = (size_t)b * CC * HW;

  const unsigned short* gHi = Ahi + (size_t)b * CC * CC;   // uniform async bases
  const unsigned short* gLo = Alo + (size_t)b * CC * CC;

  if (tid < 128) sBias[tid] = biasP[(size_t)b * CC + rowBase + tid];

  // per-thread chunk geometry for A staging: 512 x 16B chunks per plane
  const int ar0 = tid >> 2, aseg0 = tid & 3;                 // chunk tid
  const int ar1 = (tid + 256) >> 2, aseg1 = (tid + 256) & 3; // chunk tid+256

  // issue one A tile (hi+lo) as 4 async-to-LDS instructions per wave
#define STAGE_A_ASYNC(k0_, nbuf_)                                                   \
  do {                                                                              \
    const unsigned go0 = (unsigned)((((rowBase + ar0) * CC) + (k0_) + aseg0 * 8) * 2); \
    const unsigned go1 = (unsigned)((((rowBase + ar1) * CC) + (k0_) + aseg1 * 8) * 2); \
    const unsigned lh0 = lds_off_u32(&sA[(nbuf_)][0][ar0 * LDSS + aseg0 * 8]);      \
    const unsigned ll0 = lds_off_u32(&sA[(nbuf_)][1][ar0 * LDSS + aseg0 * 8]);      \
    const unsigned lh1 = lds_off_u32(&sA[(nbuf_)][0][ar1 * LDSS + aseg1 * 8]);      \
    const unsigned ll1 = lds_off_u32(&sA[(nbuf_)][1][ar1 * LDSS + aseg1 * 8]);      \
    asm volatile("global_load_async_to_lds_b128 %0, %1, %2"                         \
                 :: "v"(lh0), "v"(go0), "s"(gHi) : "memory");                       \
    asm volatile("global_load_async_to_lds_b128 %0, %1, %2"                         \
                 :: "v"(ll0), "v"(go0), "s"(gLo) : "memory");                       \
    asm volatile("global_load_async_to_lds_b128 %0, %1, %2"                         \
                 :: "v"(lh1), "v"(go1), "s"(gHi) : "memory");                       \
    asm volatile("global_load_async_to_lds_b128 %0, %1, %2"                         \
                 :: "v"(ll1), "v"(go1), "s"(gLo) : "memory");                       \
  } while (0)

  v8f acc[2][4] = {};

  STAGE_A_ASYNC(0, 0);              // prologue: tile 0 in flight
  int buf = 0;

  for (int k0 = 0; k0 < CC; k0 += 32) {
    // ---- stage B phase 1: batch-issue all 4 global float4 loads ----
    float4 v4a[4];
#pragma unroll
    for (int i = 0; i < 4; ++i) {
      const int idx = tid + i * 256;             // 0..1023 -> 32 k-rows x 32 float4
      const int c  = idx >> 5;                   // k within tile
      const int s4 = idx & 31;
      v4a[i] = *(const float4*)&x[xBatch + (size_t)(k0 + c) * HW + colBase + s4 * 4];
    }
    // ---- stage B phase 2: bf16 hi/lo split, store transposed to LDS ----
#pragma unroll
    for (int i = 0; i < 4; ++i) {
      const int idx = tid + i * 256;
      const int c  = idx >> 5;
      const int s4 = idx & 31;
      const float vv[4] = { v4a[i].x, v4a[i].y, v4a[i].z, v4a[i].w };
#pragma unroll
      for (int j = 0; j < 4; ++j) {
        const float v = vv[j];
        const unsigned short hb = f32_to_bf16_rne(v);
        const float lof = v - __uint_as_float((unsigned int)hb << 16);
        sBhi[(s4 * 4 + j) * LDSS + c] = hb;
        sBlo[(s4 * 4 + j) * LDSS + c] = f32_to_bf16_rne(lof);
      }
    }
    // issue next A tile into the other buffer, then retire the current one:
    // async loads complete in order, so asynccnt<=4 leaves only the new tile
    // in flight and guarantees tile k0 is resident in LDS.
    if (k0 + 32 < CC) {
      STAGE_A_ASYNC(k0 + 32, buf ^ 1);
      __builtin_prefetch(&x[xBatch + (size_t)(k0 + 32 + (tid >> 3)) * HW + colBase + (tid & 7) * 16], 0, 0);
      asm volatile("s_wait_asynccnt 0x4" ::: "memory");
    } else {
      asm volatile("s_wait_asynccnt 0x0" ::: "memory");
    }
    __syncthreads();

    // ---- load fragments in contiguous k order (same for A and B) ----
    // A: lane holds row M=lane%16; halfs 0..15 -> K=lg*16..lg*16+15 (contiguous)
    // B: lane holds col N=lane%16; halfs 0..15 -> K=lg*16..lg*16+15 (contiguous)
    v16bf ah[2], al[2], bh[4], bl[4];
#pragma unroll
    for (int mt = 0; mt < 2; ++mt) {
      const int row = wr * 32 + mt * 16 + lm;
      const unsigned short* p0 = &sA[buf][0][row * LDSS];
      const unsigned short* p1 = &sA[buf][1][row * LDSS];
      V16 t0, t1;
      t0.q[0] = *(const uint4*)&p0[lg * 16];  t0.q[1] = *(const uint4*)&p0[lg * 16 + 8];
      t1.q[0] = *(const uint4*)&p1[lg * 16];  t1.q[1] = *(const uint4*)&p1[lg * 16 + 8];
      ah[mt] = t0.v;  al[mt] = t1.v;
    }
#pragma unroll
    for (int nt = 0; nt < 4; ++nt) {
      const int col = wc * 64 + nt * 16 + lm;
      const unsigned short* p0 = &sBhi[col * LDSS + lg * 16];
      const unsigned short* p1 = &sBlo[col * LDSS + lg * 16];
      V16 t0, t1;
      t0.q[0] = *(const uint4*)&p0[0];  t0.q[1] = *(const uint4*)&p0[8];
      t1.q[0] = *(const uint4*)&p1[0];  t1.q[1] = *(const uint4*)&p1[8];
      bh[nt] = t0.v;  bl[nt] = t1.v;
    }
    // ---- bf16x3 WMMA: hi*hi + hi*lo + lo*hi, f32 accumulate ----
#pragma unroll
    for (int mt = 0; mt < 2; ++mt)
#pragma unroll
      for (int nt = 0; nt < 4; ++nt) {
        v8f cfr = acc[mt][nt];
        cfr = WMMA_BF16(ah[mt], bl[nt], cfr);
        cfr = WMMA_BF16(al[mt], bh[nt], cfr);
        cfr = WMMA_BF16(ah[mt], bh[nt], cfr);
        acc[mt][nt] = cfr;
      }
    __syncthreads();
    buf ^= 1;
  }
#undef STAGE_A_ASYNC

  // ---- epilogue: + bias' + residual x, nontemporal store (keep x in L2) ----
#pragma unroll
  for (int mt = 0; mt < 2; ++mt)
#pragma unroll
    for (int nt = 0; nt < 4; ++nt) {
      const int rl0 = wr * 32 + mt * 16 + lg * 8;         // C/D layout: elem r -> M=lg*8+r
      const int col = colBase + wc * 64 + nt * 16 + lm;   // N = lane%16
#pragma unroll
      for (int r = 0; r < 8; ++r) {
        const int o = rowBase + rl0 + r;
        const size_t idx = xBatch + (size_t)o * HW + col;
        const float v = acc[mt][nt][r] + sBias[rl0 + r] + x[idx];
        __builtin_nontemporal_store(v, &out[idx]);
      }
    }
}

// ---------------------------------------------------------------------------
extern "C" void kernel_launch(void* const* d_in, const int* in_sizes, int n_in,
                              void* d_out, int out_size, void* d_ws, size_t ws_size,
                              hipStream_t stream) {
  (void)in_sizes; (void)n_in; (void)out_size; (void)ws_size;
  const float* x      = (const float*)d_in[0];
  const float* norm_g = (const float*)d_in[1];
  const float* norm_b = (const float*)d_in[2];
  const float* qkv_w  = (const float*)d_in[3];
  const float* qkv_b  = (const float*)d_in[4];
  const float* proj_w = (const float*)d_in[5];
  const float* proj_b = (const float*)d_in[6];
  float* out = (float*)d_out;

  // workspace carving (all offsets 256B aligned); total ~17.9 MB
  char* ws = (char*)d_ws;
  float* mean  = (float*)(ws + 0);          // 128 f32
  float* rstd  = (float*)(ws + 512);        // 128 f32
  float* pb2   = (float*)(ws + 1024);       // 512 f32 (proj@bv)
  float* biasP = (float*)(ws + 3072);       // 16x512 f32
  float* M     = (float*)(ws + 35840);      // 512x512 f32
  unsigned short* Ahi = (unsigned short*)(ws + 1084416);   // 16x512x512 bf16
  unsigned short* Alo = (unsigned short*)(ws + 9473024);   // 16x512x512 bf16

  gn_stats_kernel<<<NB * NG, 256, 0, stream>>>(x, mean, rstd);
  make_M_kernel<<<CC / 8, 256, 0, stream>>>(qkv_w, proj_w, qkv_b, M, pb2);
  fold_M_kernel<<<dim3(CC, NB), 256, 0, stream>>>(M, mean, rstd, norm_g, norm_b,
                                                  proj_b, pb2, Ahi, Alo, biasP);
  attn_gemm_kernel<<<dim3(HW / 128, CC / 128, NB), 256, 0, stream>>>(x, Ahi, Alo, biasP, out);
}
